// GroupedQueryAttention_78692390797561
// MI455X (gfx1250) — compile-verified, matrix operand order checked by reading, not run on hardware
//
#include <hip/hip_runtime.h>
#include <hip/hip_bf16.h>
#include <cstdint>

typedef __attribute__((ext_vector_type(16))) __bf16 v16bf;
typedef __attribute__((ext_vector_type(8)))  __bf16 v8bf;
typedef __attribute__((ext_vector_type(8)))  float  v8f;
typedef __attribute__((ext_vector_type(4)))  int    v4i;

#define B_      2
#define S_      2048
#define HID_    2048
#define NH_     16
#define NKV_    4
#define HD_     128
#define GROUPS_ 4
#define HALFHD_ 64

__device__ __forceinline__ v16bf ldv16(const __bf16* p) {
    return *reinterpret_cast<const v16bf*>(p);
}

// 32B LDS fragment read as two 16B loads (16B-aligned padded layout)
__device__ __forceinline__ v16bf ldlds32(const __bf16* p) {
    v8bf lo = *reinterpret_cast<const v8bf*>(p);
    v8bf hi = *reinterpret_cast<const v8bf*>(p + 8);
    return __builtin_shufflevector(lo, hi, 0, 1, 2, 3, 4, 5, 6, 7,
                                   8, 9, 10, 11, 12, 13, 14, 15);
}

// async 16B global -> LDS copy (ASYNCcnt-tracked, per-lane addresses)
__device__ __forceinline__ void async_b128(const __bf16* g, __bf16* l) {
    __builtin_amdgcn_global_load_async_to_lds_b128((v4i*)g, (v4i*)l, 0, 0);
}

// ---------------------------------------------------------------------------
// fp32 -> bf16 conversion (4 elems / thread)
// ---------------------------------------------------------------------------
__global__ void f2bf_kernel(const float* __restrict__ src,
                            __bf16* __restrict__ dst, int n) {
    int i = (blockIdx.x * blockDim.x + threadIdx.x) * 4;
    if (i >= n) return;
    float4 v = *reinterpret_cast<const float4*>(src + i);
    dst[i + 0] = (__bf16)v.x;
    dst[i + 1] = (__bf16)v.y;
    dst[i + 2] = (__bf16)v.z;
    dst[i + 3] = (__bf16)v.w;
}

// ---------------------------------------------------------------------------
// RoPE in-place on q (b,h,s,d) and k (b,kv,s,d), pairs along d
// ---------------------------------------------------------------------------
__global__ void rope_kernel(__bf16* __restrict__ qb, __bf16* __restrict__ kb,
                            const float* __restrict__ cosb,
                            const float* __restrict__ sinb) {
    const int nq = B_ * NH_ * S_ * HALFHD_;
    const int nk = B_ * NKV_ * S_ * HALFHD_;
    int tid = blockIdx.x * blockDim.x + threadIdx.x;
    if (tid >= nq + nk) return;
    __bf16* buf = qb;
    if (tid >= nq) { tid -= nq; buf = kb; }
    int j    = tid & (HALFHD_ - 1);
    int rest = tid >> 6;
    int s    = rest & (S_ - 1);
    size_t base = ((size_t)rest) * HD_ + 2 * j;
    float x0 = (float)buf[base];
    float x1 = (float)buf[base + 1];
    float c  = cosb[s * HALFHD_ + j];
    float sn = sinb[s * HALFHD_ + j];
    buf[base]     = (__bf16)(x0 * c - x1 * sn);
    buf[base + 1] = (__bf16)(x0 * sn + x1 * c);
}

// ---------------------------------------------------------------------------
// Tiled WMMA GEMM with double-buffered async-to-LDS staging.
//   C[M,N] = A[M,K] * W[N,K]^T, bf16 operands, fp32 accumulate.
// Block: 256 thr = 8 waves; block tile 128x128; wave tile 32x64 (2x4 WMMA).
// Per K=32 stage: 2x (128x32) tiles staged via GLOBAL_LOAD_ASYNC_TO_LDS_B128.
// LDS rows padded to 40 elems (80B) for conflict-free fragment reads.
// EPI 0: QKV scatter epilogue; EPI 1: fp32 store out[row*ldo + col].
// ---------------------------------------------------------------------------
#define LDSPITCH 40

template <int EPI>
__global__ __launch_bounds__(256)
void gemm_wmma_kernel(const __bf16* __restrict__ A, const __bf16* __restrict__ W,
                      int K, int ldo,
                      __bf16* __restrict__ qb, __bf16* __restrict__ kb,
                      __bf16* __restrict__ vt, float* __restrict__ out) {
    const int lane = threadIdx.x & 31;
    const int wave = threadIdx.x >> 5;
    const int col  = lane & 15;
    const int half = lane >> 4;
    const int wm   = wave >> 1;            // 0..3
    const int wn   = wave & 1;             // 0..1
    const int mblk = blockIdx.y * 128;
    const int nblk = blockIdx.x * 128;

    __shared__ __align__(128) __bf16 lA[2][128][LDSPITCH];
    __shared__ __align__(128) __bf16 lB[2][128][LDSPITCH];

    // issue one K=32 stage: each thread moves 2x16B of A and 2x16B of W
    // (4 async instructions per wave per stage)
    auto issue = [&](int ks, int buf) {
        const int k = ks * 32;
#pragma unroll
        for (int i = 0; i < 2; i++) {
            int c   = threadIdx.x * 2 + i;    // 0..511 chunks of 8 elems
            int row = c >> 2;
            int cc  = c & 3;
            async_b128(A + (size_t)(mblk + row) * K + k + cc * 8,
                       &lA[buf][row][cc * 8]);
            async_b128(W + (size_t)(nblk + row) * K + k + cc * 8,
                       &lB[buf][row][cc * 8]);
        }
    };

    v8f acc[2][4];
#pragma unroll
    for (int i = 0; i < 2; i++)
#pragma unroll
        for (int j = 0; j < 4; j++)
            acc[i][j] = (v8f){0.f, 0.f, 0.f, 0.f, 0.f, 0.f, 0.f, 0.f};

    const int nk = K / 32;
    issue(0, 0);
    for (int ks = 0; ks < nk; ks++) {
        const int buf = ks & 1;
        if (ks + 1 < nk) {
            issue(ks + 1, buf ^ 1);
            // our own 4 next-stage asyncs may remain in flight
            asm volatile("s_wait_asynccnt 0x4" ::: "memory");
        } else {
            asm volatile("s_wait_asynccnt 0x0" ::: "memory");
        }
        __syncthreads();   // all waves' async writes for buf now visible

        v16bf a[2], b[4];
#pragma unroll
        for (int mt = 0; mt < 2; mt++)
            a[mt] = ldlds32(&lA[buf][wm * 32 + mt * 16 + col][half * 16]);
#pragma unroll
        for (int nt = 0; nt < 4; nt++)
            b[nt] = ldlds32(&lB[buf][wn * 64 + nt * 16 + col][half * 16]);
#pragma unroll
        for (int mt = 0; mt < 2; mt++)
#pragma unroll
            for (int nt = 0; nt < 4; nt++)
                acc[mt][nt] = __builtin_amdgcn_wmma_f32_16x16x32_bf16(
                    false, a[mt], false, b[nt], (short)0, acc[mt][nt],
                    false, false);

        __syncthreads();   // done reading buf before it is re-filled
    }

#pragma unroll
    for (int mt = 0; mt < 2; mt++) {
#pragma unroll
        for (int nt = 0; nt < 4; nt++) {
#pragma unroll
            for (int r = 0; r < 8; r++) {
                int   row  = mblk + wm * 32 + mt * 16 + r + 8 * half;
                int   gcol = nblk + wn * 64 + nt * 16 + col;
                float v    = acc[mt][nt][r];
                if (EPI == 0) {
                    int b_ = row >> 11;           // / S_
                    int s_ = row & (S_ - 1);
                    if (gcol < 2048) {
                        int h = gcol >> 7, d = gcol & 127;
                        qb[((size_t)(b_ * NH_ + h) * S_ + s_) * HD_ + d] = (__bf16)v;
                    } else if (gcol < 2560) {
                        int c2 = gcol - 2048;
                        int h = c2 >> 7, d = c2 & 127;
                        kb[((size_t)(b_ * NKV_ + h) * S_ + s_) * HD_ + d] = (__bf16)v;
                    } else {
                        int c2 = gcol - 2560;
                        int h = c2 >> 7, d = c2 & 127;
                        vt[((size_t)(b_ * NKV_ + h) * HD_ + d) * S_ + s_] = (__bf16)v;
                    }
                } else {
                    out[(size_t)row * ldo + gcol] = v;
                }
            }
        }
    }
}

// ---------------------------------------------------------------------------
// Flash attention: 128 thr = 4 waves; each wave: 16 queries x HD=128
// 32-key chunks: 8 WMMA (QK^T over HD) + online softmax + 8 WMMA (PV)
// ---------------------------------------------------------------------------
__global__ __launch_bounds__(128)
void flash_attn_kernel(const __bf16* __restrict__ qb,
                       const __bf16* __restrict__ kb,
                       const __bf16* __restrict__ vt,
                       __bf16* __restrict__ attn) {
    const int lane = threadIdx.x & 31;
    const int wave = threadIdx.x >> 5;
    const int col  = lane & 15;
    const int half = lane >> 4;

    const int bh  = blockIdx.x;
    const int b   = bh / NH_;
    const int h   = bh % NH_;
    const int kvh = h / GROUPS_;
    const int q0  = blockIdx.y * 64 + wave * 16;

    __shared__ __align__(64) __bf16 pl[4][16][32];

    const __bf16* qrow =
        qb + ((size_t)(b * NH_ + h) * S_ + q0 + col) * HD_ + half * 16;
    v16bf qa[4];
#pragma unroll
    for (int c = 0; c < 4; c++) qa[c] = ldv16(qrow + c * 32);

    float m[8], l[8];
#pragma unroll
    for (int r = 0; r < 8; r++) { m[r] = -1e30f; l[r] = 0.f; }
    v8f o[8];
#pragma unroll
    for (int ct = 0; ct < 8; ct++)
        o[ct] = (v8f){0.f, 0.f, 0.f, 0.f, 0.f, 0.f, 0.f, 0.f};

    const __bf16* kbp = kb + (size_t)(b * NKV_ + kvh) * S_ * HD_;
    const __bf16* vbp = vt + (size_t)(b * NKV_ + kvh) * HD_ * S_;
    const float scale = 0.08838834764831845f;   // 1/sqrt(128)
    const float L2E   = 1.44269504088896f;

    for (int kbase = 0; kbase <= q0 + 15; kbase += 32) {
        if (kbase + 32 <= q0 + 15)
            __builtin_prefetch(kbp + (size_t)(kbase + 32 + col) * HD_, 0, 0);

        // ---- scores: two 16x16 key subtiles, contract HD in 4 chunks ----
        v8f s0 = (v8f){0.f, 0.f, 0.f, 0.f, 0.f, 0.f, 0.f, 0.f};
        v8f s1 = s0;
#pragma unroll
        for (int c = 0; c < 4; c++) {
            v16bf k0 = ldv16(kbp + (size_t)(kbase + col) * HD_ + c * 32 + half * 16);
            v16bf k1 = ldv16(kbp + (size_t)(kbase + 16 + col) * HD_ + c * 32 + half * 16);
            s0 = __builtin_amdgcn_wmma_f32_16x16x32_bf16(false, qa[c], false, k0,
                                                         (short)0, s0, false, false);
            s1 = __builtin_amdgcn_wmma_f32_16x16x32_bf16(false, qa[c], false, k1,
                                                         (short)0, s1, false, false);
        }

        // ---- online softmax over rows (cross-lane within 16-lane halves) ----
        float cr[8];
#pragma unroll
        for (int r = 0; r < 8; r++) {
            int   qrowi = q0 + r + 8 * half;
            float v0 = s0[r] * scale + ((kbase + col)      > qrowi ? -1e9f : 0.f);
            float v1 = s1[r] * scale + ((kbase + 16 + col) > qrowi ? -1e9f : 0.f);
            float rm = fmaxf(v0, v1);
            rm = fmaxf(rm, __shfl_xor(rm, 1));
            rm = fmaxf(rm, __shfl_xor(rm, 2));
            rm = fmaxf(rm, __shfl_xor(rm, 4));
            rm = fmaxf(rm, __shfl_xor(rm, 8));
            float mn = fmaxf(m[r], rm);
            float c  = exp2f((m[r] - mn) * L2E);
            float p0 = exp2f((v0 - mn) * L2E);
            float p1 = exp2f((v1 - mn) * L2E);
            float rs = p0 + p1;
            rs += __shfl_xor(rs, 1);
            rs += __shfl_xor(rs, 2);
            rs += __shfl_xor(rs, 4);
            rs += __shfl_xor(rs, 8);
            l[r] = l[r] * c + rs;
            m[r] = mn;
            cr[r] = c;
            // C-layout -> LDS so we can reload in A-layout for PV
            pl[wave][r + 8 * half][col]      = (__bf16)p0;
            pl[wave][r + 8 * half][col + 16] = (__bf16)p1;
        }
#pragma unroll
        for (int ct = 0; ct < 8; ct++)
#pragma unroll
            for (int r = 0; r < 8; r++) o[ct][r] *= cr[r];

        // cross-lane LDS hazard (invisible to compiler): drain DS stores
        asm volatile("s_wait_dscnt 0" ::: "memory");
        v16bf pa = *reinterpret_cast<const v16bf*>(&pl[wave][col][half * 16]);

        // ---- PV: V^T rows are key-contiguous (vt layout (d,s)) ----
#pragma unroll
        for (int ct = 0; ct < 8; ct++) {
            v16bf vb = ldv16(vbp + (size_t)(ct * 16 + col) * S_ + kbase + half * 16);
            o[ct] = __builtin_amdgcn_wmma_f32_16x16x32_bf16(false, pa, false, vb,
                                                            (short)0, o[ct],
                                                            false, false);
        }
    }

    // ---- normalize + store to (b, s, h*HD + d) for output GEMM ----
#pragma unroll
    for (int ct = 0; ct < 8; ct++) {
#pragma unroll
        for (int r = 0; r < 8; r++) {
            int s_ = q0 + r + 8 * half;
            int d  = ct * 16 + col;
            float val = o[ct][r] / l[r];
            attn[((size_t)b * S_ + s_) * HID_ + h * HD_ + d] = (__bf16)val;
        }
    }
}

// ---------------------------------------------------------------------------
extern "C" void kernel_launch(void* const* d_in, const int* in_sizes, int n_in,
                              void* d_out, int out_size, void* d_ws, size_t ws_size,
                              hipStream_t stream) {
    const float* hs   = (const float*)d_in[0];
    const float* fcos = (const float*)d_in[1];
    const float* fsin = (const float*)d_in[2];
    // d_in[3] attention_mask unused (causal mask applied arithmetically)
    const float* Wq = (const float*)d_in[4];
    const float* Wk = (const float*)d_in[5];
    const float* Wv = (const float*)d_in[6];
    const float* Wo = (const float*)d_in[7];
    float* out = (float*)d_out;

    char* ws = (char*)d_ws;
    size_t off = 0;
    auto alloc_bf = [&](size_t elems) {
        __bf16* p = (__bf16*)(ws + off);
        off += ((elems * sizeof(__bf16) + 255) & ~(size_t)255);
        return p;
    };

    const size_t nHS   = (size_t)B_ * S_ * HID_;
    const size_t nWQKV = (size_t)(NH_ + 2 * NKV_) * HD_ * HID_;
    const size_t nWO   = (size_t)HID_ * HID_;
    const size_t nQ    = (size_t)B_ * NH_ * S_ * HD_;
    const size_t nKV   = (size_t)B_ * NKV_ * S_ * HD_;

    __bf16* hs_bf   = alloc_bf(nHS);
    __bf16* wqkv_bf = alloc_bf(nWQKV);
    __bf16* wo_bf   = alloc_bf(nWO);
    __bf16* q_buf   = alloc_bf(nQ);
    __bf16* k_buf   = alloc_bf(nKV);
    __bf16* vt_buf  = alloc_bf(nKV);
    __bf16* attn_bf = alloc_bf(nHS);

    // 1) fp32 -> bf16 conversions
    auto cvt = [&](const float* s, __bf16* d, size_t n) {
        int grid = (int)((n / 4 + 255) / 256);
        f2bf_kernel<<<grid, 256, 0, stream>>>(s, d, (int)n);
    };
    cvt(hs, hs_bf, nHS);
    cvt(Wq, wqkv_bf, (size_t)NH_ * HD_ * HID_);
    cvt(Wk, wqkv_bf + (size_t)NH_ * HD_ * HID_, (size_t)NKV_ * HD_ * HID_);
    cvt(Wv, wqkv_bf + (size_t)(NH_ + NKV_) * HD_ * HID_, (size_t)NKV_ * HD_ * HID_);
    cvt(Wo, wo_bf, nWO);

    // 2) fused QKV projection (M=4096, N=3072, K=2048), scatter epilogue
    gemm_wmma_kernel<0><<<dim3(3072 / 128, (B_ * S_) / 128), 256, 0, stream>>>(
        hs_bf, wqkv_bf, HID_, 0, q_buf, k_buf, vt_buf, nullptr);

    // 3) RoPE on q and k
    {
        int n = B_ * (NH_ + NKV_) * S_ * HALFHD_;
        rope_kernel<<<(n + 255) / 256, 256, 0, stream>>>(q_buf, k_buf, fcos, fsin);
    }

    // 4) flash attention: grid (B*NH, S/64), 4 waves x 16 queries
    flash_attn_kernel<<<dim3(B_ * NH_, S_ / 64), 128, 0, stream>>>(
        q_buf, k_buf, vt_buf, attn_bf);

    // 5) output projection (M=4096, N=2048, K=2048) -> fp32 d_out
    gemm_wmma_kernel<1><<<dim3(HID_ / 128, (B_ * S_) / 128), 256, 0, stream>>>(
        attn_bf, wo_bf, HID_, HID_, nullptr, nullptr, nullptr, out);
}
